// UserBehaviorGCN_29953101922728
// MI455X (gfx1250) — matrix operand order repaired and checked by reading, not compiled, not run on hardware
//
#include <hip/hip_runtime.h>

// ---------------- problem constants ----------------
#define NN    50000
#define EE    800000
#define DD    128
#define HH    128
#define RR    3
#define OUTD  10
#define NPAD  50048   // 391 * 128, GEMM tile padding

// ---------------- vector types ----------------
typedef __attribute__((ext_vector_type(16))) __bf16          v16bf;
typedef __attribute__((ext_vector_type(8)))  float           v8f;
typedef __attribute__((ext_vector_type(16))) unsigned short  v16s;
typedef __attribute__((ext_vector_type(8)))  unsigned short  v8s;
typedef __attribute__((ext_vector_type(4)))  unsigned short  v4s;

__device__ __forceinline__ unsigned short f2bf(float f) {
  unsigned u = __float_as_uint(f);
  unsigned r = (u + 0x7FFFu + ((u >> 16) & 1u)) >> 16;   // RNE
  return (unsigned short)r;
}
__device__ __forceinline__ float bf2f(unsigned short s) {
  return __uint_as_float(((unsigned)s) << 16);
}

// ---------------- elementwise fp32 -> bf16 ----------------
__global__ void cvt_f32_bf16(const float* __restrict__ in, unsigned short* __restrict__ out, int n) {
  int i = blockIdx.x * 256 + threadIdx.x;
  if (i < n) out[i] = f2bf(in[i]);
}

// ---------------- weight prep: fp32 [K,NC] row-major -> bf16 [NC,K] ----------------
__global__ void transpose_cvt(const float* __restrict__ in, unsigned short* __restrict__ out,
                              int K, int NC) {
  int i = blockIdx.x * 256 + threadIdx.x;
  if (i >= K * NC) return;
  int k = i / NC, n = i - k * NC;
  out[n * K + k] = f2bf(in[i]);
}

// ---------------- fill fp32 ----------------
__global__ void fillf(float* __restrict__ p, int n, float v) {
  int i = blockIdx.x * 256 + threadIdx.x;
  if (i < n) p[i] = v;
}

// ---------------- WMMA GEMM: C[M,NC] = A[M,K](bf16) * Bt[NC,K](bf16)^T ----------------
// block = 256 threads (8 waves); block owns 128 rows; wave w owns rows [w*16, w*16+16).
// Bt (whole weight matrix, <=32KB bf16) is staged into LDS once per block: all 8 waves
// share it, and per-WMMA operand fetches become short-latency ds_load_b128 pairs
// instead of L2 round trips. A fragments are preloaded to registers (no cross-wave reuse).
template<int K, int NC>
__global__ void gemm_bf16(const unsigned short* __restrict__ A,
                          const unsigned short* __restrict__ Bt,
                          float* __restrict__ Cf, unsigned short* __restrict__ Cbf, int M) {
  constexpr int NT = NC / 16;
  constexpr int KT = K / 32;
  const int lane = threadIdx.x & 31;
  const int wave = threadIdx.x >> 5;
  const int m0   = blockIdx.x * 128 + wave * 16;
  const int half = lane >> 4;          // 0: lanes with k 0-7,16-23; 1: lanes with k 8-15,24-31 (per 32-k step)
  const int l15  = lane & 15;

  // ---- stage the full transposed weight matrix into LDS (coalesced 16B chunks) ----
  __shared__ unsigned short blds[NC * K];
  {
    constexpr int CH = (NC * K) / 8;   // 16-byte chunks
#pragma unroll
    for (int c0 = 0; c0 < CH; c0 += 256) {
      int c = c0 + threadIdx.x;
      *(v8s*)(blds + (size_t)c * 8) = *(const v8s*)(Bt + (size_t)c * 8);
    }
  }

  v8f acc[NT];
#pragma unroll
  for (int i = 0; i < NT; ++i) {
    v8f z = {0.f, 0.f, 0.f, 0.f, 0.f, 0.f, 0.f, 0.f};
    acc[i] = z;
  }

  // ---- preload all A fragments (16x32 each) for the full K loop ----
  const unsigned short* arow = A + (size_t)(m0 + l15) * K + half * 8;
  v16s as[KT];
#pragma unroll
  for (int ks = 0; ks < KT; ++ks) {
    v8s alo = *(const v8s*)(arow + ks * 32);
    v8s ahi = *(const v8s*)(arow + ks * 32 + 16);
#pragma unroll
    for (int i = 0; i < 8; ++i) { as[ks][i] = alo[i]; as[ks][i + 8] = ahi[i]; }
  }

  __syncthreads();

  const unsigned short* bb = blds + (size_t)l15 * K + half * 8;
#pragma unroll
  for (int ks = 0; ks < KT; ++ks) {
    v16bf a = __builtin_bit_cast(v16bf, as[ks]);
#pragma unroll
    for (int nt = 0; nt < NT; ++nt) {
      const unsigned short* bcol = bb + nt * 16 * K + ks * 32;
      v8s b0 = *(const v8s*)(bcol);
      v8s b1 = *(const v8s*)(bcol + 16);
      v16s bs;
#pragma unroll
      for (int i = 0; i < 8; ++i) { bs[i] = b0[i]; bs[i + 8] = b1[i]; }
      acc[nt] = __builtin_amdgcn_wmma_f32_16x16x32_bf16(
          false, a, false, __builtin_bit_cast(v16bf, bs), (short)0, acc[nt], false, false);
    }
  }

  // C/D layout: VGPR r -> row m0 + r + half*8, col = nt*16 + l15
#pragma unroll
  for (int nt = 0; nt < NT; ++nt) {
#pragma unroll
    for (int r = 0; r < 8; ++r) {
      int rr = m0 + half * 8 + r;
      if (rr < M) {
        int cc = nt * 16 + l15;
        float v = acc[nt][r];
        if (Cf)  Cf[(size_t)rr * NC + cc] = v;
        if (Cbf) Cbf[(size_t)rr * NC + cc] = f2bf(v);
      }
    }
  }
}

// ---------------- edge gather + atomic aggregate (wave32 per edge) ----------------
__device__ __forceinline__ void atomicMaxF(float* a, float v) {
  int iv = __float_as_int(v);
  if (iv >= 0) atomicMax((int*)a, iv);
  else         atomicMin((unsigned int*)a, (unsigned int)iv);
}

__global__ void edge_aggr(const unsigned short* __restrict__ xw,   // [R, NPAD, 128] bf16
                          const int* __restrict__ ei,              // [2, E]
                          const int* __restrict__ et,              // [E]
                          float* __restrict__ agg,                 // [NPAD, 128] fp32
                          int E, int isMax) {
  const int lane = threadIdx.x & 31;
  int e = blockIdx.x * 8 + (threadIdx.x >> 5);
  if (e >= E) return;
  // edge id / indices are wave-uniform: force into SGPRs -> scalar index loads
  e = __builtin_amdgcn_readfirstlane(e);
  const int s = __builtin_amdgcn_readfirstlane(ei[e]);
  const int d = __builtin_amdgcn_readfirstlane(ei[E + e]);
  const int r = __builtin_amdgcn_readfirstlane(et[e]);
  // prefetch the streaming index arrays well ahead
  __builtin_prefetch(ei + e + 4096, 0, 1);
  __builtin_prefetch(et + e + 4096, 0, 1);

  const v4s m = *((const v4s*)(xw + ((size_t)r * NPAD + s) * HH) + lane);
  float* ap = agg + (size_t)d * HH + lane * 4;
  if (isMax) {
#pragma unroll
    for (int j = 0; j < 4; ++j) atomicMaxF(ap + j, bf2f(m[j]));
  } else {
#pragma unroll
    for (int j = 0; j < 4; ++j) atomicAdd(ap + j, bf2f(m[j]));
  }
}

// ---------------- row-wise (+bias +base) -> LayerNorm -> GELU (+residual) ----------------
// one wave32 per row; Hd in {128, 64}
__global__ void post_ln_gelu(const float* __restrict__ pre, const float* __restrict__ addb,
                             const float* __restrict__ bias,
                             const float* __restrict__ g, const float* __restrict__ be,
                             const float* __restrict__ resid, float rscale,
                             float* __restrict__ outf, unsigned short* __restrict__ outbf,
                             int Nrows, int Hd, int ffilter) {
  const int lane = threadIdx.x & 31;
  const int row  = blockIdx.x * 8 + (threadIdx.x >> 5);
  if (row >= Nrows) return;
  const int c = Hd >> 5;                 // elems per lane (4 or 2)
  float xv[4];
  float s = 0.f;
  for (int j = 0; j < c; ++j) {
    const int col = lane * c + j;
    float v = pre[(size_t)row * Hd + col];
    if (ffilter) v = (fabsf(v) < 3.0e38f) ? v : 0.f;   // empty segments (-inf) -> 0
    if (addb) v += addb[(size_t)row * Hd + col];
    if (bias) v += bias[col];
    xv[j] = v;
    s += v;
  }
  for (int o = 16; o > 0; o >>= 1) s += __shfl_xor(s, o, 32);
  const float mean = s / (float)Hd;
  float vs = 0.f;
  for (int j = 0; j < c; ++j) { float d0 = xv[j] - mean; vs += d0 * d0; }
  for (int o = 16; o > 0; o >>= 1) vs += __shfl_xor(vs, o, 32);
  const float inv = rsqrtf(vs / (float)Hd + 1e-5f);
  for (int j = 0; j < c; ++j) {
    const int col = lane * c + j;
    float y = (xv[j] - mean) * inv * g[col] + be[col];
    float ge = 0.5f * y * (1.f + erff(y * 0.70710678118654752f));   // exact GELU
    if (resid) ge += rscale * resid[(size_t)row * Hd + col];
    if (outf)  outf[(size_t)row * Hd + col] = ge;
    if (outbf) outbf[(size_t)row * Hd + col] = f2bf(ge);
  }
}

// ---------------- final tiny GEMM (64x10) + log_softmax, thread per node ----------------
__global__ void final_logits(const float* __restrict__ c2,   // [N,64]
                             const float* __restrict__ Wc3,  // [64,10]
                             const float* __restrict__ bc3,  // [10]
                             float* __restrict__ out, int Nn) {
  int n = blockIdx.x * 256 + threadIdx.x;
  if (n >= Nn) return;
  float l[OUTD];
#pragma unroll
  for (int j = 0; j < OUTD; ++j) l[j] = bc3[j];
  const float* row = c2 + (size_t)n * 64;
  for (int k = 0; k < 64; ++k) {
    float v = row[k];
#pragma unroll
    for (int j = 0; j < OUTD; ++j) l[j] += v * Wc3[k * OUTD + j];
  }
  float m = l[0];
#pragma unroll
  for (int j = 1; j < OUTD; ++j) m = fmaxf(m, l[j]);
  float sum = 0.f;
#pragma unroll
  for (int j = 0; j < OUTD; ++j) sum += expf(l[j] - m);
  const float lse = m + logf(sum);
#pragma unroll
  for (int j = 0; j < OUTD; ++j) out[(size_t)n * OUTD + j] = l[j] - lse;
}

// ---------------- host orchestration ----------------
extern "C" void kernel_launch(void* const* d_in, const int* in_sizes, int n_in,
                              void* d_out, int out_size, void* d_ws, size_t ws_size,
                              hipStream_t stream) {
  (void)in_sizes; (void)n_in; (void)out_size; (void)ws_size;
  const float* x     = (const float*)d_in[0];
  const int*   ei    = (const int*)d_in[1];
  const int*   et    = (const int*)d_in[2];
  const float* W1    = (const float*)d_in[3];
  const float* root1 = (const float*)d_in[4];
  const float* b1    = (const float*)d_in[5];
  const float* g1    = (const float*)d_in[6];
  const float* be1   = (const float*)d_in[7];
  const float* W2    = (const float*)d_in[8];
  const float* root2 = (const float*)d_in[9];
  const float* b2    = (const float*)d_in[10];
  const float* g2    = (const float*)d_in[11];
  const float* be2   = (const float*)d_in[12];
  const float* Wc1   = (const float*)d_in[13];
  const float* bc1   = (const float*)d_in[14];
  const float* gc1   = (const float*)d_in[15];
  const float* bec1  = (const float*)d_in[16];
  const float* Wc2   = (const float*)d_in[17];
  const float* bc2   = (const float*)d_in[18];
  const float* gc2   = (const float*)d_in[19];
  const float* bec2  = (const float*)d_in[20];
  const float* Wc3   = (const float*)d_in[21];
  const float* bc3   = (const float*)d_in[22];
  float* out = (float*)d_out;

  // workspace carve-up (256B aligned)
  char* ws = (char*)d_ws;
  size_t off = 0;
  auto carve = [&](size_t bytes) { char* p = ws + off; off = (off + bytes + 255) & ~(size_t)255; return p; };
  unsigned short* bfA   = (unsigned short*)carve((size_t)NPAD * HH * 2);      // shared A-operand: x/h1/h/c1 in bf16
  unsigned short* xw    = (unsigned short*)carve((size_t)RR * NPAD * HH * 2); // per-relation transforms (bf16)
  float*          agg   = (float*)carve((size_t)NPAD * HH * 4);               // aggregation / c2pre / c2
  float*          base  = (float*)carve((size_t)NPAD * HH * 4);               // root-transform / c1pre
  float*          h1    = (float*)carve((size_t)NPAD * HH * 4);               // residual
  unsigned short* W1t   = (unsigned short*)carve((size_t)RR * DD * HH * 2);
  unsigned short* r1t   = (unsigned short*)carve((size_t)DD * HH * 2);
  unsigned short* W2t   = (unsigned short*)carve((size_t)RR * HH * HH * 2);
  unsigned short* r2t   = (unsigned short*)carve((size_t)HH * HH * 2);
  unsigned short* Wc1t  = (unsigned short*)carve((size_t)HH * HH * 2);
  unsigned short* Wc2t  = (unsigned short*)carve((size_t)HH * 64 * 2);

  const dim3 B256(256);
  const int GT128 = (128 * 128 + 255) / 256;     // 16384-elem transposes
  const int GEMMG = NPAD / 128;                  // 391 row tiles
  const int FILLG = (NPAD * HH + 255) / 256;
  const int POSTG = (NN + 7) / 8;
  const int EDGEG = (EE + 7) / 8;

  // ---- weight prep (bf16 + transpose) ----
  for (int r = 0; r < RR; ++r)
    transpose_cvt<<<GT128, B256, 0, stream>>>(W1 + (size_t)r * DD * HH, W1t + (size_t)r * DD * HH, DD, HH);
  transpose_cvt<<<GT128, B256, 0, stream>>>(root1, r1t, DD, HH);
  for (int r = 0; r < RR; ++r)
    transpose_cvt<<<GT128, B256, 0, stream>>>(W2 + (size_t)r * HH * HH, W2t + (size_t)r * HH * HH, HH, HH);
  transpose_cvt<<<GT128, B256, 0, stream>>>(root2, r2t, HH, HH);
  transpose_cvt<<<GT128, B256, 0, stream>>>(Wc1, Wc1t, HH, HH);
  transpose_cvt<<<(128 * 64 + 255) / 256, B256, 0, stream>>>(Wc2, Wc2t, HH, 64);

  // ---- layer 1 ----
  cvt_f32_bf16<<<(NN * DD + 255) / 256, B256, 0, stream>>>(x, bfA, NN * DD);
  for (int r = 0; r < RR; ++r)
    gemm_bf16<128, 128><<<GEMMG, B256, 0, stream>>>(bfA, W1t + (size_t)r * DD * HH,
                                                    nullptr, xw + (size_t)r * NPAD * HH, NN);
  gemm_bf16<128, 128><<<GEMMG, B256, 0, stream>>>(bfA, r1t, base, nullptr, NN);
  fillf<<<FILLG, B256, 0, stream>>>(agg, NPAD * HH, -__builtin_inff());
  edge_aggr<<<EDGEG, B256, 0, stream>>>(xw, ei, et, agg, EE, 1);
  post_ln_gelu<<<POSTG, B256, 0, stream>>>(agg, base, b1, g1, be1, nullptr, 0.f,
                                           h1, bfA, NN, HH, 1);

  // ---- layer 2 ----
  for (int r = 0; r < RR; ++r)
    gemm_bf16<128, 128><<<GEMMG, B256, 0, stream>>>(bfA, W2t + (size_t)r * HH * HH,
                                                    nullptr, xw + (size_t)r * NPAD * HH, NN);
  gemm_bf16<128, 128><<<GEMMG, B256, 0, stream>>>(bfA, r2t, base, nullptr, NN);
  fillf<<<FILLG, B256, 0, stream>>>(agg, NPAD * HH, 0.f);
  edge_aggr<<<EDGEG, B256, 0, stream>>>(xw, ei, et, agg, EE, 0);
  post_ln_gelu<<<POSTG, B256, 0, stream>>>(agg, base, b2, g2, be2, h1, 0.5f,
                                           nullptr, bfA, NN, HH, 0);

  // ---- classifier ----
  gemm_bf16<128, 128><<<GEMMG, B256, 0, stream>>>(bfA, Wc1t, base, nullptr, NN);
  post_ln_gelu<<<POSTG, B256, 0, stream>>>(base, nullptr, bc1, gc1, bec1, nullptr, 0.f,
                                           nullptr, bfA, NN, HH, 0);
  gemm_bf16<128, 64><<<GEMMG, B256, 0, stream>>>(bfA, Wc2t, agg, nullptr, NN);
  post_ln_gelu<<<POSTG, B256, 0, stream>>>(agg, nullptr, bc2, gc2, bec2, nullptr, 0.f,
                                           agg, nullptr, NN, 64, 0);
  final_logits<<<(NN + 255) / 256, B256, 0, stream>>>(agg, Wc3, bc3, out, NN);
}
